// ContextAugHead_82892868813363
// MI455X (gfx1250) — matrix operand order repaired and checked by reading, not compiled, not run on hardware
//
#include <hip/hip_runtime.h>
#include <hip/hip_bf16.h>
#include <math.h>

// ---------------------------------------------------------------------------
// Problem constants (fixed by the reference)
// ---------------------------------------------------------------------------
#define BB 8
#define SS 2048
#define HH 768
#define NROWS (BB * SS)        // 16384
#define NGRP 256               // indicator in [0,256)
#define NCLS 3

typedef _Float16 v16h __attribute__((ext_vector_type(16)));
typedef _Float16 v8h  __attribute__((ext_vector_type(8)));
typedef float    v8f  __attribute__((ext_vector_type(8)));

// ---------------------------------------------------------------------------
// Kernel 0: zero the tiny reduction buffer (2 parities x 16 floats)
// ---------------------------------------------------------------------------
__global__ void ca_init_red(float* red) {
    if (threadIdx.x < 32) red[threadIdx.x] = 0.0f;
}

// ---------------------------------------------------------------------------
// Kernel 1: per-(batch, group) sums + counts.
// One block per (b,g): build the member list in LDS, then each thread
// accumulates 3 of the 768 dims over the member rows. Every input row is
// read exactly once across its group's block.
// ---------------------------------------------------------------------------
__global__ void ca_groupsum(const float* __restrict__ x,
                            const int*   __restrict__ ind,
                            float*       __restrict__ sums,
                            unsigned*    __restrict__ counts) {
    const int b = blockIdx.x >> 8;
    const int g = blockIdx.x & 255;
    __shared__ int list[SS];
    __shared__ int cnt;
    if (threadIdx.x == 0) cnt = 0;
    __syncthreads();
    const int* indb = ind + b * SS;
    for (int j = threadIdx.x; j < SS; j += 256) {
        if (indb[j] == g) { int p = atomicAdd(&cnt, 1); list[p] = j; }
    }
    __syncthreads();
    const int n = cnt;
    const float* xb = x + (size_t)b * SS * HH;
    float a0 = 0.f, a1 = 0.f, a2 = 0.f;
    for (int ii = 0; ii < n; ++ii) {
        const float* row = xb + (size_t)list[ii] * HH;
        a0 += row[threadIdx.x];
        a1 += row[threadIdx.x + 256];
        a2 += row[threadIdx.x + 512];
    }
    float* s = sums + (size_t)blockIdx.x * HH;
    s[threadIdx.x]       = a0;
    s[threadIdx.x + 256] = a1;
    s[threadIdx.x + 512] = a2;
    if (threadIdx.x == 0) counts[blockIdx.x] = (unsigned)n;
}

// ---------------------------------------------------------------------------
// Kernel 2: pre-swizzle Wu into WMMA B-fragment order (f16).
// B[k][n] = Wu[n][k]. Fragment: tile (kt,ot), lane l holds col n = ot*16+(l&15),
// slot s (0..15) holds K = kt*32 + 8*(l>>4) + s + 8*(s>>3). One f16 per thread.
// ---------------------------------------------------------------------------
__global__ void ca_packB(const float* __restrict__ Wu, _Float16* __restrict__ Bp) {
    const int idx  = blockIdx.x * 256 + threadIdx.x;   // 24*48*512 = 589824
    const int s    = idx & 15;
    const int lane = (idx >> 4) & 31;
    const int ot   = (idx >> 9) % 48;
    const int kt   = idx / (512 * 48);
    const int o = ot * 16 + (lane & 15);
    const int k = kt * 32 + 8 * (lane >> 4) + s + ((s >> 3) << 3);
    Bp[idx] = (_Float16)Wu[o * HH + k];
}

// ---------------------------------------------------------------------------
// Kernel 3: gather cell_states rows (group mean) into f16 GEMM operand A.
// ---------------------------------------------------------------------------
__global__ void ca_gather(const int*      __restrict__ ind,
                          const float*    __restrict__ sums,
                          const unsigned* __restrict__ counts,
                          _Float16*       __restrict__ cells) {
    const size_t idx = (size_t)blockIdx.x * 256 + threadIdx.x; // NROWS*HH
    const int d   = (int)(idx % HH);
    const int row = (int)(idx / HH);
    const int b = row >> 11, s = row & 2047;
    const int bg = (b << 8) + ind[b * SS + s];
    const float inv = 1.0f / (float)counts[bg];
    cells[idx] = (_Float16)(sums[(size_t)bg * HH + d] * inv);
}

// ---------------------------------------------------------------------------
// Kernel 4: main GEMM  h = tanh(cells @ Wu^T + bu), f16 WMMA, f32 accum.
// Wave computes a 32(M) x 64(N) tile: 2 A fragments, 4 held B fragments,
// 8 accumulators -> 24 B/lane of L1 traffic per WMMA. All 12 loads of a
// k-iteration are issued before the first WMMA so the scheduler can retire
// them with decreasing s_wait_loadcnt values instead of full drains.
//   A frag: two contiguous 16B loads/lane (ISA 16-bit A 16x32 layout).
//   B frag: one contiguous 32B load/lane from the pre-swizzled buffer.
// Epilogue uses the f32 C/D layout (VGPR r, lane -> row r+8*(l>>4), col l&15)
// to fuse +bu and tanh, storing h in f16.
// ---------------------------------------------------------------------------
__global__ void ca_gemm(const _Float16* __restrict__ cells,
                        const _Float16* __restrict__ Bp,
                        const float*    __restrict__ bu,
                        _Float16*       __restrict__ hbuf) {
    const int lane = threadIdx.x & 31;
    const int gw   = blockIdx.x * 8 + (threadIdx.x >> 5); // 6144 waves
    const int mtile = gw / 12;            // 0..511, 32 rows each
    const int ngrp  = gw % 12;            // 0..11, 64 cols each
    const int row0 = mtile * 32;
    const int n0   = ngrp * 64;
    const int m  = lane & 15;
    const int hs = lane >> 4;

    const _Float16* Arow0 = cells + (size_t)(row0 + m) * HH + 8 * hs;
    const _Float16* Arow1 = Arow0 + (size_t)16 * HH;

    v8f acc[2][4];
    const v8f zero = {0.f, 0.f, 0.f, 0.f, 0.f, 0.f, 0.f, 0.f};
#pragma unroll
    for (int u = 0; u < 2; ++u)
#pragma unroll
        for (int t = 0; t < 4; ++t) acc[u][t] = zero;

    for (int k0 = 0; k0 < HH; k0 += 32) {
        // ---- issue ALL loads for this k-step first ----
        const v8h lo0 = *(const v8h*)(Arow0 + k0);
        const v8h hi0 = *(const v8h*)(Arow0 + k0 + 16);
        const v8h lo1 = *(const v8h*)(Arow1 + k0);
        const v8h hi1 = *(const v8h*)(Arow1 + k0 + 16);
        const _Float16* bbase =
            Bp + ((size_t)(k0 >> 5) * 48 + (n0 >> 4)) * 512 + lane * 16;
        v16h barr[4];
#pragma unroll
        for (int t = 0; t < 4; ++t)
            barr[t] = *(const v16h*)(bbase + (size_t)t * 512);

        v16h a0, a1;
#pragma unroll
        for (int i = 0; i < 8; ++i) {
            a0[i] = lo0[i]; a0[8 + i] = hi0[i];
            a1[i] = lo1[i]; a1[8 + i] = hi1[i];
        }

        // ---- 8 WMMAs consume the fragments ----
#pragma unroll
        for (int t = 0; t < 4; ++t) {
            acc[0][t] = __builtin_amdgcn_wmma_f32_16x16x32_f16(
                false, a0, false, barr[t], (short)0, acc[0][t], false, false);
            acc[1][t] = __builtin_amdgcn_wmma_f32_16x16x32_f16(
                false, a1, false, barr[t], (short)0, acc[1][t], false, false);
        }
    }

#pragma unroll
    for (int u = 0; u < 2; ++u) {
#pragma unroll
        for (int t = 0; t < 4; ++t) {
            const int n = n0 + t * 16 + (lane & 15);
            const float bn = bu[n];
#pragma unroll
            for (int r = 0; r < 8; ++r) {
                const int row = row0 + u * 16 + r + 8 * hs;
                const float hv = tanhf(acc[u][t][r] + bn);
                hbuf[(size_t)row * HH + n] = (_Float16)hv;
            }
        }
    }
}

// ---------------------------------------------------------------------------
// Kernel 5: per-row head. One wave per row: logits = h @ Wp^T + bp (Wp in LDS),
// shuffle-reduce, then lane 0 does log-sum-exp CE, argmax, and atomically
// accumulates weighted-loss sums and tp/fp/fn split by row parity
// (even rows = "sep", odd rows = "tok").
// red layout per parity (16 floats): [0]=sum_w*nll [1]=sum_w
//                                    [2..4]=tp [5..7]=fp [8..10]=fn
// ---------------------------------------------------------------------------
__global__ void ca_head(const _Float16* __restrict__ hbuf,
                        const int*      __restrict__ labels,
                        const float*    __restrict__ Wp,
                        const float*    __restrict__ bp,
                        const float*    __restrict__ lw,
                        float*          __restrict__ red) {
    __shared__ float sWp[NCLS * HH];
    for (int i = threadIdx.x; i < NCLS * HH; i += 256) sWp[i] = Wp[i];
    __syncthreads();

    const int lane = threadIdx.x & 31;
    const int row  = blockIdx.x * 8 + (threadIdx.x >> 5);
    const _Float16* h = hbuf + (size_t)row * HH;

    float a0 = 0.f, a1 = 0.f, a2 = 0.f;
#pragma unroll 4
    for (int j = 0; j < HH / 32; ++j) {
        const int d = j * 32 + lane;
        const float v = (float)h[d];
        a0 += v * sWp[d];
        a1 += v * sWp[HH + d];
        a2 += v * sWp[2 * HH + d];
    }
    for (int off = 16; off > 0; off >>= 1) {
        a0 += __shfl_down(a0, off, 32);
        a1 += __shfl_down(a1, off, 32);
        a2 += __shfl_down(a2, off, 32);
    }
    if (lane == 0) {
        const float l0 = a0 + bp[0], l1 = a1 + bp[1], l2 = a2 + bp[2];
        const int label = labels[row];
        int pred = 0; float best = l0;
        if (l1 > best) { best = l1; pred = 1; }
        if (l2 > best) { best = l2; pred = 2; }
        const float mx  = fmaxf(l0, fmaxf(l1, l2));
        const float lse = mx + logf(expf(l0 - mx) + expf(l1 - mx) + expf(l2 - mx));
        const float ll  = (label == 0) ? l0 : ((label == 1) ? l1 : l2);
        const float nll = lse - ll;
        const float w   = lw[label];
        float* r = red + (row & 1) * 16;
        atomicAdd(r + 0, w * nll);
        atomicAdd(r + 1, w);
        if (pred == label) {
            atomicAdd(r + 2 + pred, 1.0f);
        } else {
            atomicAdd(r + 5 + pred, 1.0f);
            atomicAdd(r + 8 + label, 1.0f);
        }
    }
}

// ---------------------------------------------------------------------------
// Kernel 6: finalize. Output = flat pytree of
// ((sep_loss, {0..2: {f1,fn,fp,precision,recall,tp}}), (tok_loss, {...}))
// -> 38 floats (inner dict keys alphabetical).
// ---------------------------------------------------------------------------
__global__ void ca_finalize(const float* __restrict__ red, float* __restrict__ out) {
    if (threadIdx.x != 0 || blockIdx.x != 0) return;
    for (int p = 0; p < 2; ++p) {
        const float* r = red + p * 16;
        const int base = p * 19;
        out[base] = r[0] / r[1];                       // loss
        for (int t = 0; t < NCLS; ++t) {
            const float tp = r[2 + t], fp = r[5 + t], fn = r[8 + t];
            const float prec = tp / (tp + fp);
            const float rec  = tp / (tp + fn);
            const float f1   = 2.0f * prec * rec / (prec + rec);
            float* o = out + base + 1 + t * 6;
            o[0] = f1; o[1] = fn; o[2] = fp; o[3] = prec; o[4] = rec; o[5] = tp;
        }
    }
}

// ---------------------------------------------------------------------------
// Launcher. Workspace layout (bytes, all 256-aligned):
//   sums   f32 [8*256*768]          @ 0          (6,291,456)
//   counts u32 [8*256]              @ 6,291,456  (8,192)
//   cells  f16 [16384*768]          @ 6,299,648  (25,165,824)
//   Bpack  f16 [24*48*512]          @ 31,465,472 (1,179,648)
//   hbuf   f16 [16384*768]          @ 32,645,120 (25,165,824)
//   red    f32 [32]                 @ 57,810,944 (128)
// ---------------------------------------------------------------------------
extern "C" void kernel_launch(void* const* d_in, const int* in_sizes, int n_in,
                              void* d_out, int out_size, void* d_ws, size_t ws_size,
                              hipStream_t stream) {
    const float* x   = (const float*)d_in[0];   // encoded_states [8,2048,768]
    const int*   ind = (const int*)d_in[1];     // indicator      [8,2048]
    const int*   lab = (const int*)d_in[2];     // ca_label       [8,2048]
    const float* Wu  = (const float*)d_in[3];   // [768,768]
    const float* bu  = (const float*)d_in[4];   // [768]
    const float* Wp  = (const float*)d_in[5];   // [3,768]
    const float* bp  = (const float*)d_in[6];   // [3]
    const float* lw  = (const float*)d_in[7];   // [3]

    char* ws = (char*)d_ws;
    float*     sums   = (float*)(ws + 0);
    unsigned*  counts = (unsigned*)(ws + 6291456);
    _Float16*  cells  = (_Float16*)(ws + 6299648);
    _Float16*  Bpack  = (_Float16*)(ws + 31465472);
    _Float16*  hbuf   = (_Float16*)(ws + 32645120);
    float*     red    = (float*)(ws + 57810944);

    ca_init_red<<<1, 32, 0, stream>>>(red);
    ca_groupsum<<<BB * NGRP, 256, 0, stream>>>(x, ind, sums, counts);
    ca_packB<<<(24 * 48 * 512) / 256, 256, 0, stream>>>(Wu, Bpack);
    ca_gather<<<(NROWS * HH) / 256, 256, 0, stream>>>(ind, sums, counts, cells);
    ca_gemm<<<(NROWS / 32) * 12 / 8, 256, 0, stream>>>(cells, Bpack, bu, hbuf);
    ca_head<<<NROWS / 8, 256, 0, stream>>>(hbuf, lab, Wp, bp, lw, red);
    ca_finalize<<<1, 1, 0, stream>>>(red, (float*)d_out);
}
